// AttnBlock_52072183497096
// MI455X (gfx1250) — compile-verified, matrix operand order checked by reading, not run on hardware
//
#include <hip/hip_runtime.h>
#include <math.h>

// Problem constants (from reference): x [4, 512, 64, 64], 32 groups.
#define B_   4
#define C_   512
#define HW_  4096
#define G_   32
#define CG_  16
#define NI_  (HW_ / 16)   // 256 row blocks of 16 tokens
#define NC_  (C_ / 64)    // 8 col groups of 64 channels

typedef __attribute__((ext_vector_type(16))) _Float16 v16h;
typedef __attribute__((ext_vector_type(8)))  float    v8f;

// ---------------------------------------------------------------------------
// CDNA5 helpers.
// Generic pointers to LDS carry the LDS byte offset in their low 32 bits
// (ISA 10.2), so truncation gives the VDST operand
// for the async-to-LDS instructions.
// ---------------------------------------------------------------------------
__device__ __forceinline__ unsigned ldsoff(const void* p) {
    return (unsigned)(size_t)p;
}

// Per-lane async copy: 32 lanes x 16B = 512B per instruction, tracked on
// ASYNCcnt (not LOADcnt), completes in order.
__device__ __forceinline__ void async_copy16(unsigned lds, const void* g) {
    asm volatile("global_load_async_to_lds_b128 %0, %1, off"
                 :: "v"(lds), "v"(g) : "memory");
}
__device__ __forceinline__ void wait_async_le16() {
    asm volatile("s_wait_asynccnt 0x10" ::: "memory");
}
__device__ __forceinline__ void wait_async_0() {
    asm volatile("s_wait_asynccnt 0x0" ::: "memory");
}

__device__ __forceinline__ v8f wmma16(v16h a, v16h b, v8f c) {
    // (neg_a, A, neg_b, B, c_mod, C, reuse_a, reuse_b)
    return __builtin_amdgcn_wmma_f32_16x16x32_f16(false, a, false, b,
                                                  (short)0, c, false, false);
}

// A/B fragment from global, row-major with K contiguous (A: lane&15 = row,
// B stored [N][K]: lane&15 = col; lane>>4 selects the K half).
__device__ __forceinline__ v16h load_frag_g(const _Float16* __restrict__ base,
                                            int row0, int k0, int ld, int lane) {
    return *(const v16h*)(base + (size_t)(row0 + (lane & 15)) * ld + k0 + (lane >> 4) * 16);
}

// ---------------------------------------------------------------------------
// 1) fp32 -> f16 weight convert
// ---------------------------------------------------------------------------
__global__ void cvt_f16_kernel(const float* __restrict__ in,
                               _Float16* __restrict__ out, int n) {
    int i = blockIdx.x * blockDim.x + threadIdx.x;
    if (i < n) out[i] = (_Float16)in[i];
}

// ---------------------------------------------------------------------------
// 2) GroupNorm statistics: one block per (b, group); each group's data is a
//    contiguous 16ch x 4096px = 256KB span of x -> coalesced reduce.
// ---------------------------------------------------------------------------
__global__ void __launch_bounds__(256)
gn_stats_kernel(const float* __restrict__ x,
                float* __restrict__ mean, float* __restrict__ rstd) {
    const int bg = blockIdx.x;                      // b*32 + g
    const float* p = x + (size_t)bg * (CG_ * HW_);
    const int N = CG_ * HW_;                        // 65536
    float s = 0.f, ss = 0.f;
    for (int i = threadIdx.x; i < N; i += 256) {
        float v = p[i];
        s += v; ss += v * v;
    }
    __shared__ float r1[256], r2[256];
    r1[threadIdx.x] = s; r2[threadIdx.x] = ss;
    __syncthreads();
    for (int k = 128; k > 0; k >>= 1) {
        if (threadIdx.x < k) {
            r1[threadIdx.x] += r1[threadIdx.x + k];
            r2[threadIdx.x] += r2[threadIdx.x + k];
        }
        __syncthreads();
    }
    if (threadIdx.x == 0) {
        float m = r1[0] / (float)N;
        float var = r2[0] / (float)N - m * m;
        mean[bg] = m;
        rstd[bg] = rsqrtf(var + 1e-6f);
    }
}

// ---------------------------------------------------------------------------
// 3) GN apply + transpose to token-major f16: hnT[b][i][c].
// ---------------------------------------------------------------------------
__global__ void __launch_bounds__(256)
gn_apply_kernel(const float* __restrict__ x,
                const float* __restrict__ mean, const float* __restrict__ rstd,
                const float* __restrict__ gns, const float* __restrict__ gnb,
                _Float16* __restrict__ hnT) {
    size_t idx = (size_t)blockIdx.x * blockDim.x + threadIdx.x;  // [b][i][c]
    int c = (int)(idx & (C_ - 1));
    int i = (int)((idx >> 9) & (HW_ - 1));
    int b = (int)(idx >> 21);
    int gidx = b * G_ + (c >> 4);
    float v = x[((size_t)b * C_ + c) * HW_ + i];
    hnT[idx] = (_Float16)((v - mean[gidx]) * rstd[gidx] * gns[c] + gnb[c]);
}

// ---------------------------------------------------------------------------
// 4) QKV GEMM. Block = 4 waves sharing one 16-token A strip of hnT, staged
//    once into LDS via async-to-LDS. Waves take 4 different 64-channel output
//    strips; weight fragments double-buffered from global (L2 resident).
//    mode 0: token-major out [b][i][n] (qT,kT); mode 1: channel-major (v).
// ---------------------------------------------------------------------------
#define LDA_ (C_ + 8)   // padded A row stride in halves: 16B bank step
__global__ void __launch_bounds__(128)
gemm_qkv_kernel(const _Float16* __restrict__ hnT,
                const _Float16* __restrict__ W16,
                const float* __restrict__ bias,
                _Float16* __restrict__ out, int mode) {
    __shared__ _Float16 sA[16 * LDA_];
    const int wid = threadIdx.x >> 5, lane = threadIdx.x & 31;
    const int half = blockIdx.x & 1;
    const int i16  = (blockIdx.x >> 1) & (NI_ - 1);
    const int b    = blockIdx.x >> 9;
    const int i0 = i16 * 16, c0 = (half * 4 + wid) * 64;
    const _Float16* Ag = hnT + ((size_t)b * HW_ + i0) * C_;

    // cooperative async stage of the 16x512 A strip (32 x 512B)
#pragma unroll
    for (int s = wid; s < 32; s += 4) {
        int row = s >> 1, h = s & 1;
        async_copy16(ldsoff(&sA[row * LDA_ + h * 256 + lane * 8]),
                     Ag + (size_t)row * C_ + h * 256 + lane * 8);
    }
    wait_async_0();
    __syncthreads();

    v8f zero = {};
    v8f acc[4] = {zero, zero, zero, zero};
    v16h b0[4], b1[4];
#pragma unroll
    for (int t = 0; t < 4; ++t) b0[t] = load_frag_g(W16, c0 + t * 16, 0, C_, lane);
#pragma unroll
    for (int kk = 0; kk < 16; ++kk) {
        v16h* cur = (kk & 1) ? b1 : b0;
        v16h* nxt = (kk & 1) ? b0 : b1;
        if (kk + 1 < 16) {
#pragma unroll
            for (int t = 0; t < 4; ++t)
                nxt[t] = load_frag_g(W16, c0 + t * 16, (kk + 1) * 32, C_, lane);
        }
        v16h a = *(const v16h*)&sA[(lane & 15) * LDA_ + kk * 32 + (lane >> 4) * 16];
#pragma unroll
        for (int t = 0; t < 4; ++t) acc[t] = wmma16(a, cur[t], acc[t]);
    }
    const int col = lane & 15, rb = (lane >> 4) * 8;
#pragma unroll
    for (int t = 0; t < 4; ++t) {
        float bv = bias[c0 + t * 16 + col];
#pragma unroll
        for (int r = 0; r < 8; ++r) {
            float v = acc[t][r] + bv;
            if (mode == 0)
                out[(size_t)b * HW_ * C_ + (size_t)(i0 + rb + r) * C_ + c0 + t * 16 + col] = (_Float16)v;
            else
                out[(size_t)b * C_ * HW_ + (size_t)(c0 + t * 16 + col) * HW_ + i0 + rb + r] = (_Float16)v;
        }
    }
}

// ---------------------------------------------------------------------------
// 5) Scores. Block = 4 waves (4 consecutive 16-token row blocks, same batch).
//    Per-wave Q strip staged in LDS once; the block-shared K^T stream is
//    async-staged in double-buffered 32KB chunks (64 j-rows x 256 k-halves).
//    Hot loop: ds_load fragments + WMMA only; s_wait_asynccnt 16 pipelining.
// ---------------------------------------------------------------------------
#define SCHK_ 256            // staged K halves per chunk
#define SLDB_ (SCHK_ + 8)    // padded row stride
__global__ void __launch_bounds__(128)
attn_scores_kernel(const _Float16* __restrict__ qT,
                   const _Float16* __restrict__ kT,
                   _Float16* __restrict__ S,
                   float* __restrict__ rowmax) {
    __shared__ _Float16 sA[4][16 * LDA_];     // per-wave q strips (66.5KB)
    __shared__ _Float16 sB[2][64 * SLDB_];    // k chunks (67.6KB)
    const int wid = threadIdx.x >> 5, lane = threadIdx.x & 31;
    const int quad = blockIdx.x & 63;
    const int b    = blockIdx.x >> 6;
    const int i0 = (quad * 4 + wid) * 16;
    const _Float16* q = qT + (size_t)b * HW_ * C_;
    const _Float16* kp = kT + (size_t)b * HW_ * C_;
    _Float16* Sb = S + (size_t)b * HW_ * HW_;
    const float scale = 0.044194173824159216f;   // 512^-0.5
    const int col = lane & 15, rb = (lane >> 4) * 8;

    // per-wave Q strip stage (32 x 512B)
#pragma unroll
    for (int s = 0; s < 32; ++s) {
        int row = s >> 1, h = s & 1;
        async_copy16(ldsoff(&sA[wid][row * LDA_ + h * 256 + lane * 8]),
                     q + (size_t)(i0 + row) * C_ + h * 256 + lane * 8);
    }
    // chunk stage: it = jgroup*2 + khalf; 64 rows x 512B, 16 instr/wave
    auto stageB = [&](int it) {
        const _Float16* src = kp + (size_t)((it >> 1) * 64) * C_ + (it & 1) * SCHK_;
        _Float16* dst = &sB[it & 1][0];
#pragma unroll
        for (int s = wid; s < 64; s += 4)
            async_copy16(ldsoff(dst + s * SLDB_ + lane * 8),
                         src + (size_t)s * C_ + lane * 8);
    };
    stageB(0);

    float rmax[8];
#pragma unroll
    for (int r = 0; r < 8; ++r) rmax[r] = -1e30f;
    v8f zero = {};
    v8f acc[4] = {zero, zero, zero, zero};

    const int TOTAL = NI_ * 2;                 // 512 chunk iterations
    for (int it = 0; it < TOTAL; ++it) {
        if (it + 1 < TOTAL) { stageB(it + 1); wait_async_le16(); }
        else wait_async_0();
        __syncthreads();
        const _Float16* bb = &sB[it & 1][0];
#pragma unroll
        for (int s = 0; s < 8; ++s) {
            int k0 = (it & 1) * SCHK_ + s * 32;
            v16h a = *(const v16h*)&sA[wid][(lane & 15) * LDA_ + k0 + (lane >> 4) * 16];
#pragma unroll
            for (int t = 0; t < 4; ++t) {
                v16h bf = *(const v16h*)(bb + (size_t)(t * 16 + col) * SLDB_ + s * 32 + (lane >> 4) * 16);
                acc[t] = wmma16(a, bf, acc[t]);
            }
        }
        __syncthreads();
        if (it & 1) {                          // j-group complete
            int j0 = (it >> 1) * 64;
#pragma unroll
            for (int t = 0; t < 4; ++t) {
#pragma unroll
                for (int r = 0; r < 8; ++r) {
                    float sv = acc[t][r] * scale;
                    rmax[r] = fmaxf(rmax[r], sv);
                    Sb[(size_t)(i0 + rb + r) * HW_ + j0 + t * 16 + col] = (_Float16)sv;
                }
                acc[t] = zero;
            }
        }
    }
#pragma unroll
    for (int m = 1; m < 16; m <<= 1)
#pragma unroll
        for (int r = 0; r < 8; ++r)
            rmax[r] = fmaxf(rmax[r], __shfl_xor(rmax[r], m, 32));
    if ((lane & 15) == 0)
#pragma unroll
        for (int r = 0; r < 8; ++r)
            rowmax[(size_t)b * HW_ + i0 + rb + r] = rmax[r];
}

// ---------------------------------------------------------------------------
// 6) Softmax row pass: P = exp(S - rowmax); rowinv = 1/sum (folded into O).
// ---------------------------------------------------------------------------
__global__ void __launch_bounds__(256)
attn_softmax_kernel(_Float16* __restrict__ S,
                    const float* __restrict__ rowmax,
                    float* __restrict__ rowinv) {
    const int row = blockIdx.x;                    // b*HW + i
    _Float16* Sr = S + (size_t)row * HW_;
    const float m = rowmax[row];
    __shared__ float red[256];
    float sum = 0.f;
    for (int j = threadIdx.x; j < HW_; j += 256) {
        float p = __expf((float)Sr[j] - m);
        Sr[j] = (_Float16)p;
        sum += p;
    }
    red[threadIdx.x] = sum;
    __syncthreads();
    for (int k = 128; k > 0; k >>= 1) {
        if (threadIdx.x < k) red[threadIdx.x] += red[threadIdx.x + k];
        __syncthreads();
    }
    if (threadIdx.x == 0) rowinv[row] = 1.0f / red[0];
}

// ---------------------------------------------------------------------------
// 7) O = P * V^T (K = j = 4096), row-scaled by 1/sum. Block = 4 waves with the
//    SAME 64-channel V strip (staged in LDS, shared) and 4 different 16-token
//    P strips (double-buffered global fragments).
// ---------------------------------------------------------------------------
__global__ void __launch_bounds__(128)
attn_out_kernel(const _Float16* __restrict__ P,
                const _Float16* __restrict__ v,
                const float* __restrict__ rowinv,
                _Float16* __restrict__ hT) {
    __shared__ _Float16 sB[2][64 * SLDB_];    // v chunks (67.6KB)
    const int wid = threadIdx.x >> 5, lane = threadIdx.x & 31;
    const int c64  = blockIdx.x & 7;
    const int quad = (blockIdx.x >> 3) & 63;
    const int b    = blockIdx.x >> 9;
    const int i0 = (quad * 4 + wid) * 16, c0 = c64 * 64;
    const _Float16* Pb = P + ((size_t)b * HW_ + i0) * HW_;
    const _Float16* vb = v + ((size_t)b * C_ + c0) * HW_;

    auto stageB = [&](int it) {               // chunk it: j in [it*256, it*256+256)
        const _Float16* src = vb + it * SCHK_;
        _Float16* dst = &sB[it & 1][0];
#pragma unroll
        for (int s = wid; s < 64; s += 4)
            async_copy16(ldsoff(dst + s * SLDB_ + lane * 8),
                         src + (size_t)s * HW_ + lane * 8);
    };
    stageB(0);

    v8f zero = {};
    v8f acc[4] = {zero, zero, zero, zero};
    v16h a_cur = *(const v16h*)(Pb + (size_t)(lane & 15) * HW_ + (lane >> 4) * 16);
    const int NCHUNK = HW_ / SCHK_;           // 16
    for (int it = 0; it < NCHUNK; ++it) {
        if (it + 1 < NCHUNK) { stageB(it + 1); wait_async_le16(); }
        else wait_async_0();
        __syncthreads();
        const _Float16* bb = &sB[it & 1][0];
#pragma unroll
        for (int s = 0; s < 8; ++s) {
            int jn = it * SCHK_ + (s + 1) * 32;          // prefetch next P frag
            v16h a_nxt = a_cur;
            if (jn < HW_)
                a_nxt = *(const v16h*)(Pb + (size_t)(lane & 15) * HW_ + jn + (lane >> 4) * 16);
#pragma unroll
            for (int t = 0; t < 4; ++t) {
                v16h bf = *(const v16h*)(bb + (size_t)(t * 16 + (lane & 15)) * SLDB_ + s * 32 + (lane >> 4) * 16);
                acc[t] = wmma16(a_cur, bf, acc[t]);
            }
            a_cur = a_nxt;
        }
        __syncthreads();
    }
    const int col = lane & 15, rb = (lane >> 4) * 8;
#pragma unroll
    for (int r = 0; r < 8; ++r) {
        float inv = rowinv[(size_t)b * HW_ + i0 + rb + r];
#pragma unroll
        for (int t = 0; t < 4; ++t)
            hT[(size_t)b * HW_ * C_ + (size_t)(i0 + rb + r) * C_ + c0 + t * 16 + col] =
                (_Float16)(acc[t][r] * inv);
    }
}

// ---------------------------------------------------------------------------
// 8) Projection + bias + residual, f32 transposed store to d_out [b][c][hw].
//    Same LDS-staged-A structure as the QKV GEMM.
// ---------------------------------------------------------------------------
__global__ void __launch_bounds__(128)
proj_kernel(const _Float16* __restrict__ hT,
            const _Float16* __restrict__ Wp16,
            const float* __restrict__ bproj,
            const float* __restrict__ x,
            float* __restrict__ out) {
    __shared__ _Float16 sA[16 * LDA_];
    const int wid = threadIdx.x >> 5, lane = threadIdx.x & 31;
    const int half = blockIdx.x & 1;
    const int i16  = (blockIdx.x >> 1) & (NI_ - 1);
    const int b    = blockIdx.x >> 9;
    const int i0 = i16 * 16, c0 = (half * 4 + wid) * 64;
    const _Float16* Ag = hT + ((size_t)b * HW_ + i0) * C_;

#pragma unroll
    for (int s = wid; s < 32; s += 4) {
        int row = s >> 1, h = s & 1;
        async_copy16(ldsoff(&sA[row * LDA_ + h * 256 + lane * 8]),
                     Ag + (size_t)row * C_ + h * 256 + lane * 8);
    }
    wait_async_0();
    __syncthreads();

    v8f zero = {};
    v8f acc[4] = {zero, zero, zero, zero};
    v16h b0[4], b1[4];
#pragma unroll
    for (int t = 0; t < 4; ++t) b0[t] = load_frag_g(Wp16, c0 + t * 16, 0, C_, lane);
#pragma unroll
    for (int kk = 0; kk < 16; ++kk) {
        v16h* cur = (kk & 1) ? b1 : b0;
        v16h* nxt = (kk & 1) ? b0 : b1;
        if (kk + 1 < 16) {
#pragma unroll
            for (int t = 0; t < 4; ++t)
                nxt[t] = load_frag_g(Wp16, c0 + t * 16, (kk + 1) * 32, C_, lane);
        }
        v16h a = *(const v16h*)&sA[(lane & 15) * LDA_ + kk * 32 + (lane >> 4) * 16];
#pragma unroll
        for (int t = 0; t < 4; ++t) acc[t] = wmma16(a, cur[t], acc[t]);
    }
    const int col = lane & 15, rb = (lane >> 4) * 8;
#pragma unroll
    for (int t = 0; t < 4; ++t) {
        int c = c0 + t * 16 + col;
        float bv = bproj[c];
#pragma unroll
        for (int r = 0; r < 8; ++r) {
            size_t oidx = ((size_t)b * C_ + c) * HW_ + i0 + rb + r;
            out[oidx] = acc[t][r] + bv + x[oidx];
        }
    }
}

// ---------------------------------------------------------------------------
// Host-side orchestration (graph-capture safe: launches only).
// ---------------------------------------------------------------------------
extern "C" void kernel_launch(void* const* d_in, const int* in_sizes, int n_in,
                              void* d_out, int out_size, void* d_ws, size_t ws_size,
                              hipStream_t stream) {
    const float* x   = (const float*)d_in[0];
    const float* gns = (const float*)d_in[1];
    const float* gnb = (const float*)d_in[2];
    const float* wq  = (const float*)d_in[3];
    const float* bq  = (const float*)d_in[4];
    const float* wk  = (const float*)d_in[5];
    const float* bk  = (const float*)d_in[6];
    const float* wv  = (const float*)d_in[7];
    const float* bv  = (const float*)d_in[8];
    const float* wp  = (const float*)d_in[9];
    const float* bp  = (const float*)d_in[10];
    float* out = (float*)d_out;

    char* ws = (char*)d_ws;
    size_t off = 0;
    auto take = [&](size_t bytes) -> char* {
        char* p = ws + off;
        off = (off + bytes + 255) & ~(size_t)255;
        return p;
    };
    const size_t act = (size_t)B_ * HW_ * C_ * sizeof(_Float16);  // 16 MB
    _Float16* hnT  = (_Float16*)take(act);
    _Float16* qT   = (_Float16*)take(act);
    _Float16* kT   = (_Float16*)take(act);
    _Float16* vC   = (_Float16*)take(act);
    _Float16* hT   = (_Float16*)take(act);
    _Float16* wq16 = (_Float16*)take((size_t)C_ * C_ * 2);
    _Float16* wk16 = (_Float16*)take((size_t)C_ * C_ * 2);
    _Float16* wv16 = (_Float16*)take((size_t)C_ * C_ * 2);
    _Float16* wp16 = (_Float16*)take((size_t)C_ * C_ * 2);
    float* mean    = (float*)take((size_t)B_ * G_ * 4);
    float* rstd    = (float*)take((size_t)B_ * G_ * 4);
    float* rowmax  = (float*)take((size_t)B_ * HW_ * 4);
    float* rowinv  = (float*)take((size_t)B_ * HW_ * 4);
    _Float16* P    = (_Float16*)take((size_t)B_ * HW_ * HW_ * 2); // 128 MB

    const int wn = C_ * C_;
    cvt_f16_kernel<<<wn / 256, 256, 0, stream>>>(wq, wq16, wn);
    cvt_f16_kernel<<<wn / 256, 256, 0, stream>>>(wk, wk16, wn);
    cvt_f16_kernel<<<wn / 256, 256, 0, stream>>>(wv, wv16, wn);
    cvt_f16_kernel<<<wn / 256, 256, 0, stream>>>(wp, wp16, wn);

    gn_stats_kernel<<<B_ * G_, 256, 0, stream>>>(x, mean, rstd);
    gn_apply_kernel<<<(B_ * HW_ * C_) / 256, 256, 0, stream>>>(x, mean, rstd, gns, gnb, hnT);

    const int gemm_blocks = B_ * NI_ * 2;    // 2048: (b, i16, half-of-8-cgroups)
    gemm_qkv_kernel<<<gemm_blocks, 128, 0, stream>>>(hnT, wq16, bq, qT, 0);
    gemm_qkv_kernel<<<gemm_blocks, 128, 0, stream>>>(hnT, wk16, bk, kT, 0);
    gemm_qkv_kernel<<<gemm_blocks, 128, 0, stream>>>(hnT, wv16, bv, vC, 1);

    attn_scores_kernel<<<B_ * 64, 128, 0, stream>>>(qT, kT, P, rowmax);
    attn_softmax_kernel<<<B_ * HW_, 256, 0, stream>>>(P, rowmax, rowinv);
    attn_out_kernel<<<B_ * 64 * NC_, 128, 0, stream>>>(P, vC, rowinv, hT);

    proj_kernel<<<gemm_blocks, 128, 0, stream>>>(hT, wp16, bp, x, out);
}
